// HeteroGAT_5961414607309
// MI455X (gfx1250) — matrix operand order repaired and checked by reading, not compiled, not run on hardware
//
#include <hip/hip_runtime.h>
#include <hip/hip_bf16.h>

typedef __attribute__((ext_vector_type(16))) __bf16 v16bf;
typedef __attribute__((ext_vector_type(8)))  float  v8f;

#define N_NODES   20000
#define NE        100000
#define IN_DIM    32
#define HID       96
#define GHEADS    2
#define GD        48
#define HHEADS    6
#define HD        16
#define NEG       0.2f
#define MT        (N_NODES/16)   // 1250 (exact)
#define NT_HID    (HID/16)       // 6
#define KT_HID    (HID/32)       // 3
#define KT_IN     (IN_DIM/32)    // 1
#define WAVES_PER_BLOCK 8

static inline int divup(int a,int b){return (a+b-1)/b;}

// ---------------- device helpers ----------------
__device__ inline float lrelu(float x){ return x>0.f ? x : NEG*x; }

__device__ inline unsigned short f2bf(float f){
  unsigned u = __float_as_uint(f);
  unsigned r = u + 0x7FFFu + ((u>>16)&1u);   // round-to-nearest-even
  return (unsigned short)(r>>16);
}
// order-preserving float->uint key for atomicMax-based segment max
__device__ inline unsigned fenc(float f){
  unsigned u=__float_as_uint(f);
  return (u&0x80000000u) ? ~u : (u|0x80000000u);
}
__device__ inline float seg_max(const unsigned* mk,int i){
  unsigned k=mk[i];
  unsigned u=(k&0x80000000u)?(k^0x80000000u):~k;
  if((u&0x7F800000u)==0x7F800000u) return 0.f;   // empty/-inf -> 0 (ref semantics)
  return __uint_as_float(u);
}

// ---------------- packing into WMMA fragment layout ----------------
// A 16x32 bf16 frag (8 VGPR): lane L: m=L&15, kbase=(L>>4)*8
//   j<8 : K = kbase+j ;  j>=8 : K = 16+kbase+(j-8)
__global__ void pack_a_kernel(const float* __restrict__ A, unsigned short* __restrict__ out,
                              int Mtiles,int Ktiles,int lda){
  int tid = blockIdx.x*blockDim.x + threadIdx.x;
  int total = Mtiles*Ktiles*32;
  if(tid>=total) return;
  int lane = tid & 31, frag = tid >> 5;
  int kt = frag % Ktiles, mt = frag / Ktiles;
  int m = mt*16 + (lane&15);
  int kbase = (lane>>4)*8;
  unsigned short* o = out + (size_t)frag*512 + lane*16;
  const float* a = A + (size_t)m*lda + kt*32;
  #pragma unroll
  for(int j=0;j<16;++j){
    int k = (j<8) ? (kbase+j) : (16+kbase+(j-8));
    o[j] = f2bf(a[k]);
  }
}
// B 32x16 bf16 frag: lane L: n=L&15, K = kt*32 + (L>>4)*8 + (j<8 ? j : 16+(j-8))
__global__ void pack_b_kernel(const float* __restrict__ B, unsigned short* __restrict__ out,
                              int Ntiles,int Ktiles,int ldb){
  int tid = blockIdx.x*blockDim.x + threadIdx.x;
  int total = Ntiles*Ktiles*32;
  if(tid>=total) return;
  int lane = tid & 31, frag = tid >> 5;
  int kt = frag % Ktiles, nt = frag / Ktiles;
  int n = nt*16 + (lane&15);
  int kb = kt*32 + (lane>>4)*8;
  unsigned short* o = out + (size_t)frag*512 + lane*16;
  #pragma unroll
  for(int j=0;j<16;++j) o[j] = f2bf(B[(size_t)(kb + ((j<8) ? j : (16+(j-8))))*ldb + n]);
}

// ---------------- bf16 WMMA GEMM ----------------
// One wave computes a full 16-row strip (all 6 N-tiles): A fragments loaded once,
// packed-B panel (<=18KB) staged in LDS per block and re-read via ds_load.
// 18 v_wmma per wave (Kt=3 x Nt=6) between stores.
__global__ void gemm_bf16_kernel(const unsigned short* __restrict__ Ap,
                                 const unsigned short* __restrict__ Bp,
                                 const float* __restrict__ bias,
                                 float* __restrict__ C,
                                 int Mtiles,int Ntiles,int Ktiles,int ldc,int accumulate){
  __shared__ __align__(32) unsigned short bsh[NT_HID*KT_HID*512];  // max 18KB
  // stage B panel: Ntiles*Ktiles*512 ushorts, 16B vectors
  int nv = Ntiles*Ktiles*64;   // uint4 count
  const uint4* bsrc = (const uint4*)Bp;
  uint4* bdst = (uint4*)bsh;
  for(int i=threadIdx.x;i<nv;i+=blockDim.x) bdst[i] = bsrc[i];
  __syncthreads();

  int wave = (blockIdx.x*blockDim.x + threadIdx.x) >> 5;
  int lane = threadIdx.x & 31;
  if(wave >= Mtiles) return;
  int mt = wave;

  v8f acc0={},acc1={},acc2={},acc3={},acc4={},acc5={};
  const unsigned short* ap = Ap + (size_t)mt*Ktiles*512 + lane*16;
  const unsigned short* bs = bsh + lane*16;
  for(int kt=0; kt<Ktiles; ++kt){
    v16bf a = *(const v16bf*)ap; ap += 512;
    v16bf b0 = *(const v16bf*)(bs + (0*Ktiles+kt)*512);
    v16bf b1 = *(const v16bf*)(bs + (1*Ktiles+kt)*512);
    v16bf b2 = *(const v16bf*)(bs + (2*Ktiles+kt)*512);
    v16bf b3 = *(const v16bf*)(bs + (3*Ktiles+kt)*512);
    v16bf b4 = *(const v16bf*)(bs + (4*Ktiles+kt)*512);
    v16bf b5 = *(const v16bf*)(bs + (5*Ktiles+kt)*512);
    acc0 = __builtin_amdgcn_wmma_f32_16x16x32_bf16(false,a,false,b0,(short)0,acc0,false,false);
    acc1 = __builtin_amdgcn_wmma_f32_16x16x32_bf16(false,a,false,b1,(short)0,acc1,false,false);
    acc2 = __builtin_amdgcn_wmma_f32_16x16x32_bf16(false,a,false,b2,(short)0,acc2,false,false);
    acc3 = __builtin_amdgcn_wmma_f32_16x16x32_bf16(false,a,false,b3,(short)0,acc3,false,false);
    acc4 = __builtin_amdgcn_wmma_f32_16x16x32_bf16(false,a,false,b4,(short)0,acc4,false,false);
    acc5 = __builtin_amdgcn_wmma_f32_16x16x32_bf16(false,a,false,b5,(short)0,acc5,false,false);
  }
  int nl = lane&15;
  int m0 = mt*16 + ((lane>>4)<<3);
  v8f* accs[6] = {&acc0,&acc1,&acc2,&acc3,&acc4,&acc5};
  #pragma unroll
  for(int nt=0;nt<6;++nt){
    if(nt>=Ntiles) break;
    int n = nt*16 + nl;
    float bv = bias ? bias[n] : 0.f;
    v8f& acc = *accs[nt];
    #pragma unroll
    for(int r=0;r<8;++r){
      size_t idx = (size_t)(m0+r)*ldc + n;
      float v = acc[r] + bv;
      if(accumulate) C[idx] += v; else C[idx] = v;
    }
  }
}

// ---------------- utility kernels ----------------
__global__ void fill_f32_kernel(float* p,int n,float v){
  int i=blockIdx.x*blockDim.x+threadIdx.x; if(i<n) p[i]=v;
}
__global__ void fill_u32_kernel(unsigned* p,int n,unsigned v){
  int i=blockIdx.x*blockDim.x+threadIdx.x; if(i<n) p[i]=v;
}
__global__ void bias_add_kernel(float* out,const float* b,int n,int cdim){
  int i=blockIdx.x*blockDim.x+threadIdx.x; if(i<n) out[i]+=b[i%cdim];
}
__global__ void gelu_kernel(float* p,int n){
  int i=blockIdx.x*blockDim.x+threadIdx.x;
  if(i<n){ float x=p[i]; p[i]=0.5f*x*(1.f+erff(x*0.70710678118f)); }
}

// ---------------- BatchNorm ----------------
__global__ void bn_stats_kernel(const float* __restrict__ x,float* mu,float* rstd,int n,int cdim){
  __shared__ float s1[256], s2[256];
  int c = blockIdx.x;
  float a=0.f,b=0.f;
  for(int i=threadIdx.x;i<n;i+=blockDim.x){ float v=x[(size_t)i*cdim+c]; a+=v; b+=v*v; }
  s1[threadIdx.x]=a; s2[threadIdx.x]=b; __syncthreads();
  for(int st=128;st>0;st>>=1){
    if((int)threadIdx.x<st){ s1[threadIdx.x]+=s1[threadIdx.x+st]; s2[threadIdx.x]+=s2[threadIdx.x+st]; }
    __syncthreads();
  }
  if(threadIdx.x==0){
    float m=s1[0]/n; float var=s2[0]/n - m*m;
    mu[c]=m; rstd[c]=rsqrtf(var+1e-5f);
  }
}
// out = maybe_lrelu(addA + gamma*(x-mu)*rstd + beta + addB)
__global__ void bn_norm_kernel(const float* __restrict__ x,const float* mu,const float* rstd,
                               const float* gamma,const float* beta,
                               const float* __restrict__ addA,const float* __restrict__ addB,
                               float* __restrict__ out,int n,int cdim,int do_lrelu){
  int i=blockIdx.x*blockDim.x+threadIdx.x; if(i>=n) return;
  int c=i%cdim;
  float v = gamma[c]*(x[i]-mu[c])*rstd[c] + beta[c];
  if(addA) v += addA[i];
  if(addB) v += addB[i];
  out[i] = do_lrelu ? lrelu(v) : v;
}

// ---------------- GAT edge kernels (softmax over incoming edges) ----------------
__global__ void gat_attn_kernel(const float* __restrict__ hp,const float* __restrict__ att,
                                float* asrc,float* adst){
  int tid=blockIdx.x*blockDim.x+threadIdx.x;
  if(tid>=N_NODES*GHEADS) return;
  int n=tid/GHEADS, h=tid%GHEADS;
  const float* hv = hp + (size_t)n*HID + h*GD;
  const float* a0 = att + h*GD;
  const float* a1 = att + GHEADS*GD + h*GD;
  float s0=0.f,s1=0.f;
  for(int d=0;d<GD;++d){ float v=hv[d]; s0+=v*a0[d]; s1+=v*a1[d]; }
  asrc[tid]=s0; adst[tid]=s1;
}
__global__ void gat_edge_max_kernel(const int* src,const int* dst,const float* asrc,const float* adst,
                                    unsigned* mk,int ne_real,int ntot){
  int e=blockIdx.x*blockDim.x+threadIdx.x; if(e>=ntot) return;
  int s_,d_;
  if(e<ne_real){ s_=src[e]; d_=dst[e]; } else { s_=d_=e-ne_real; }   // self loops
  #pragma unroll
  for(int h=0;h<GHEADS;++h){
    float l = lrelu(asrc[s_*GHEADS+h]+adst[d_*GHEADS+h]);
    atomicMax(&mk[d_*GHEADS+h], fenc(l));
  }
}
__global__ void gat_edge_sum_kernel(const int* src,const int* dst,const float* asrc,const float* adst,
                                    const unsigned* mk,float* ssum,int ne_real,int ntot){
  int e=blockIdx.x*blockDim.x+threadIdx.x; if(e>=ntot) return;
  int s_,d_;
  if(e<ne_real){ s_=src[e]; d_=dst[e]; } else { s_=d_=e-ne_real; }
  #pragma unroll
  for(int h=0;h<GHEADS;++h){
    float l = lrelu(asrc[s_*GHEADS+h]+adst[d_*GHEADS+h]);
    atomicAdd(&ssum[d_*GHEADS+h], expf(l - seg_max(mk,d_*GHEADS+h)));
  }
}
// one wave per edge: 96 channels, 3 per lane; atomic accumulate into out[dst]
__global__ void gat_edge_aggr_kernel(const int* src,const int* dst,const float* asrc,const float* adst,
                                     const unsigned* mk,const float* ssum,
                                     const float* __restrict__ hp,float* __restrict__ gout,
                                     int ne_real,int ntot){
  int wave=(blockIdx.x*blockDim.x+threadIdx.x)>>5;
  int lane=threadIdx.x&31;
  if(wave>=ntot) return;
  int s_,d_;
  if(wave<ne_real){ s_=src[wave]; d_=dst[wave]; } else { s_=d_=wave-ne_real; }
  float alpha[GHEADS];
  #pragma unroll
  for(int h=0;h<GHEADS;++h){
    float l = lrelu(asrc[s_*GHEADS+h]+adst[d_*GHEADS+h]);
    alpha[h] = expf(l - seg_max(mk,d_*GHEADS+h)) / ssum[d_*GHEADS+h];
  }
  const float* hr = hp + (size_t)s_*HID;
  float* orow = gout + (size_t)d_*HID;
  #pragma unroll
  for(int t=0;t<3;++t){
    int dch = lane + 32*t;
    atomicAdd(&orow[dch], alpha[dch/GD]*hr[dch]);
  }
}

// ---------------- HGT kernels ----------------
// blockdiag 96x96 from [6,16,16] per-head matrices (lets per-head 16x16 maps reuse the WMMA GEMM)
__global__ void blockdiag_kernel(const float* __restrict__ rel,float* __restrict__ W){
  int i=blockIdx.x*blockDim.x+threadIdx.x;
  if(i>=HID*HID) return;
  int row=i/HID, col=i%HID;
  int hr=row>>4, hc=col>>4;
  W[i] = (hr==hc) ? rel[((size_t)hr*16+(row&15))*16 + (col&15)] : 0.f;
}
__global__ void hgt_logit_kernel(const int* src,const int* dst,
                                 const float* __restrict__ q,const float* __restrict__ ka,
                                 const float* __restrict__ p,
                                 float* __restrict__ lbuf,unsigned* mk){
  int tid=blockIdx.x*blockDim.x+threadIdx.x;
  if(tid>=NE*HHEADS) return;
  int e=tid/HHEADS, h=tid%HHEADS;
  int s_=src[e], d_=dst[e];
  const float* qv = q  + (size_t)d_*HID + h*HD;
  const float* kv = ka + (size_t)s_*HID + h*HD;
  float acc=0.f;
  #pragma unroll
  for(int d=0;d<HD;++d) acc += qv[d]*kv[d];
  float l = acc * p[h] * 0.25f;   // 1/sqrt(16)
  lbuf[tid]=l;
  atomicMax(&mk[d_*HHEADS+h], fenc(l));
}
__global__ void hgt_sum_kernel(const int* __restrict__ ei,const float* __restrict__ lbuf,
                               const unsigned* mk,float* ssum){
  int tid=blockIdx.x*blockDim.x+threadIdx.x;
  if(tid>=3*NE*HHEADS) return;
  int r = tid/(NE*HHEADS);
  int rem = tid - r*NE*HHEADS;
  int e = rem/HHEADS, h = rem%HHEADS;
  int d_ = ei[(size_t)r*2*NE + NE + e];
  atomicAdd(&ssum[d_*HHEADS+h], expf(lbuf[tid] - seg_max(mk,d_*HHEADS+h)));
}
__global__ void hgt_aggr_kernel(const int* src,const int* dst,
                                const float* __restrict__ lbuf,const unsigned* mk,const float* ssum,
                                const float* __restrict__ vm,float* __restrict__ agg){
  int wave=(blockIdx.x*blockDim.x+threadIdx.x)>>5;
  int lane=threadIdx.x&31;
  if(wave>=NE) return;
  int s_=src[wave], d_=dst[wave];
  const float* vr = vm + (size_t)s_*HID;
  float* orow = agg + (size_t)d_*HID;
  #pragma unroll
  for(int t=0;t<3;++t){
    int dch = lane + 32*t;
    int h = dch>>4;
    float alpha = expf(lbuf[(size_t)wave*HHEADS+h] - seg_max(mk,d_*HHEADS+h)) / ssum[d_*HHEADS+h];
    atomicAdd(&orow[dch], alpha*vr[dch]);
  }
}
__global__ void hgt_combine_kernel(const float* o,const float* h,const float* skip,int idx,
                                   float* res,int n){
  int i=blockIdx.x*blockDim.x+threadIdx.x; if(i>=n) return;
  float sk = 1.f/(1.f+expf(-skip[idx]));
  res[i] = sk*o[i] + (1.f-sk)*h[i];
}
__global__ void lin_kernel(const float* __restrict__ h,const float* __restrict__ w,
                           const float* b,float* out){
  int n=blockIdx.x*blockDim.x+threadIdx.x; if(n>=N_NODES) return;
  const float* hr = h + (size_t)n*HID;
  float a=0.f;
  #pragma unroll
  for(int d=0;d<HID;++d) a += hr[d]*w[d];
  out[n] = a + b[0];
}

// ---------------- host orchestration ----------------
extern "C" void kernel_launch(void* const* d_in, const int* in_sizes, int n_in,
                              void* d_out, int out_size, void* d_ws, size_t ws_size,
                              hipStream_t stream){
  (void)in_sizes;(void)n_in;(void)out_size;(void)ws_size;
  const float* x        =(const float*)d_in[0];
  const int*   ei       =(const int*)  d_in[1];
  const float* gat0_W   =(const float*)d_in[2];
  const float* gat0_att =(const float*)d_in[3];
  const float* gat0_b   =(const float*)d_in[4];
  const float* gat_W    =(const float*)d_in[5];
  const float* gat_att  =(const float*)d_in[6];
  const float* gat_b    =(const float*)d_in[7];
  const float* bn_gamma =(const float*)d_in[8];
  const float* bn_beta  =(const float*)d_in[9];
  const float* hkqv_W   =(const float*)d_in[10];
  const float* hkqv_b   =(const float*)d_in[11];
  const float* ha_rel   =(const float*)d_in[12];
  const float* hm_rel   =(const float*)d_in[13];
  const float* hp_rel   =(const float*)d_in[14];
  const float* hout_W   =(const float*)d_in[15];
  const float* hout_b   =(const float*)d_in[16];
  const float* hskip    =(const float*)d_in[17];
  const float* proj_W   =(const float*)d_in[18];
  const float* proj_b   =(const float*)d_in[19];
  const float* lin_W    =(const float*)d_in[20];
  const float* lin_b    =(const float*)d_in[21];

  // workspace bump allocator (256B aligned)
  char* base=(char*)d_ws; size_t off=0;
  auto alloc=[&](size_t bytes)->void*{
    off=(off+255)&~(size_t)255; void* p=base+off; off+=bytes; return p;
  };
  const size_t NH = (size_t)N_NODES*HID;
  unsigned short* xpack  =(unsigned short*)alloc((size_t)MT*KT_IN *512*2);
  unsigned short* apackH =(unsigned short*)alloc((size_t)MT*KT_HID*512*2);
  unsigned short* apackK =(unsigned short*)alloc((size_t)MT*KT_HID*512*2);
  unsigned short* apackV =(unsigned short*)alloc((size_t)MT*KT_HID*512*2);
  unsigned short* bpack  =(unsigned short*)alloc((size_t)NT_HID*KT_HID*512*2);
  float* wtmp  =(float*)alloc((size_t)HID*HID*4);
  float* hcur  =(float*)alloc(NH*4);
  float* gout  =(float*)alloc(NH*4);           // also HGT agg
  float* hpbuf =(float*)alloc(NH*4);           // also HGT ka
  float* injbuf=(float*)alloc(NH*4);           // also HGT k
  float* qbuf  =(float*)alloc(NH*4);
  float* vbuf  =(float*)alloc(NH*4);
  float* vmbuf =(float*)alloc(NH*4);           // also HGT result
  float* obuf  =(float*)alloc(NH*4);
  float* asrc  =(float*)alloc((size_t)N_NODES*GHEADS*4);
  float* adst  =(float*)alloc((size_t)N_NODES*GHEADS*4);
  unsigned* mk =(unsigned*)alloc((size_t)N_NODES*HHEADS*4);
  float* ssum  =(float*)alloc((size_t)N_NODES*HHEADS*4);
  float* lbuf  =(float*)alloc((size_t)3*NE*HHEADS*4);
  float* mu    =(float*)alloc(HID*4);
  float* rstd  =(float*)alloc(HID*4);

  auto fillf=[&](float* p,size_t n,float v){
    fill_f32_kernel<<<divup((int)n,256),256,0,stream>>>(p,(int)n,v); };
  auto fillu=[&](unsigned* p,size_t n,unsigned v){
    fill_u32_kernel<<<divup((int)n,256),256,0,stream>>>(p,(int)n,v); };
  auto packA=[&](const float* src,unsigned short* dst,int Kt,int lda){
    int tot=MT*Kt*32;
    pack_a_kernel<<<divup(tot,256),256,0,stream>>>(src,dst,MT,Kt,lda); };
  auto gemm=[&](const unsigned short* Ap,const float* W,const float* bias,float* C,int Kt,int Nc){
    int Nt=Nc/16; int totb=Nt*Kt*32;
    pack_b_kernel<<<divup(totb,256),256,0,stream>>>(W,bpack,Nt,Kt,Nc);
    gemm_bf16_kernel<<<divup(MT,WAVES_PER_BLOCK),WAVES_PER_BLOCK*32,0,stream>>>(
        Ap,bpack,bias,C,MT,Nt,Kt,Nc,0); };
  auto bn=[&](const float* xin,int idx,const float* addA,const float* addB,float* out,int lr){
    bn_stats_kernel<<<HID,256,0,stream>>>(xin,mu,rstd,N_NODES,HID);
    bn_norm_kernel<<<divup((int)NH,256),256,0,stream>>>(xin,mu,rstd,
        bn_gamma+(size_t)idx*HID,bn_beta+(size_t)idx*HID,addA,addB,out,(int)NH,HID,lr); };

  auto gat_layer=[&](const unsigned short* Ap,int Kt,const float* W3,const float* att3,
                     const float* b3,float* out){
    fillf(out,NH,0.f);
    for(int r=0;r<3;++r){
      const float* Wr   = W3  + (size_t)r*(Kt*32)*HID;
      const float* attr = att3+ (size_t)r*2*GHEADS*GD;
      const float* br   = b3  + (size_t)r*HID;
      gemm(Ap,Wr,nullptr,hpbuf,Kt,HID);
      gat_attn_kernel<<<divup(N_NODES*GHEADS,256),256,0,stream>>>(hpbuf,attr,asrc,adst);
      fillu(mk,(size_t)N_NODES*GHEADS,0u);
      fillf(ssum,(size_t)N_NODES*GHEADS,0.f);
      const int* srcp = ei + (size_t)r*2*NE;
      const int* dstp = srcp + NE;
      int ntot = NE + N_NODES;  // + self loops
      gat_edge_max_kernel <<<divup(ntot,256),256,0,stream>>>(srcp,dstp,asrc,adst,mk,NE,ntot);
      gat_edge_sum_kernel <<<divup(ntot,256),256,0,stream>>>(srcp,dstp,asrc,adst,mk,ssum,NE,ntot);
      gat_edge_aggr_kernel<<<divup(ntot*32,256),256,0,stream>>>(srcp,dstp,asrc,adst,mk,ssum,hpbuf,out,NE,ntot);
      bias_add_kernel<<<divup((int)NH,256),256,0,stream>>>(out,br,(int)NH,HID);
    }
  };

  auto hgt=[&](int idx,float* res){
    packA(hcur,apackH,KT_HID,HID);
    const float* kW = hkqv_W + (size_t)idx*3*HID*HID;
    const float* kb = hkqv_b + (size_t)idx*3*HID;
    gemm(apackH,kW,              kb,        injbuf,KT_HID,HID);  // k
    gemm(apackH,kW+  HID*HID,    kb+HID,    qbuf,  KT_HID,HID);  // q
    gemm(apackH,kW+2*HID*HID,    kb+2*HID,  vbuf,  KT_HID,HID);  // v
    packA(injbuf,apackK,KT_HID,HID);
    packA(vbuf,  apackV,KT_HID,HID);
    fillu(mk,(size_t)N_NODES*HHEADS,0u);
    fillf(ssum,(size_t)N_NODES*HHEADS,0.f);
    fillf(gout,NH,0.f);  // agg
    for(int r=0;r<3;++r){
      blockdiag_kernel<<<divup(HID*HID,256),256,0,stream>>>(
          ha_rel+((size_t)idx*3+r)*HHEADS*HD*HD,wtmp);
      gemm(apackK,wtmp,nullptr,hpbuf,KT_HID,HID);   // ka = k @ blockdiag(a_rel)
      const int* srcp = ei + (size_t)r*2*NE; const int* dstp = srcp+NE;
      hgt_logit_kernel<<<divup(NE*HHEADS,256),256,0,stream>>>(srcp,dstp,qbuf,hpbuf,
          hp_rel+((size_t)idx*3+r)*HHEADS, lbuf+(size_t)r*NE*HHEADS, mk);
    }
    hgt_sum_kernel<<<divup(3*NE*HHEADS,256),256,0,stream>>>(ei,lbuf,mk,ssum);
    for(int r=0;r<3;++r){
      blockdiag_kernel<<<divup(HID*HID,256),256,0,stream>>>(
          hm_rel+((size_t)idx*3+r)*HHEADS*HD*HD,wtmp);
      gemm(apackV,wtmp,nullptr,vmbuf,KT_HID,HID);   // vm = v @ blockdiag(m_rel)
      const int* srcp = ei + (size_t)r*2*NE; const int* dstp = srcp+NE;
      hgt_aggr_kernel<<<divup(NE*32,256),256,0,stream>>>(srcp,dstp,
          lbuf+(size_t)r*NE*HHEADS,mk,ssum,vmbuf,gout);
    }
    gelu_kernel<<<divup((int)NH,256),256,0,stream>>>(gout,(int)NH);
    packA(gout,apackH,KT_HID,HID);
    gemm(apackH,hout_W+(size_t)idx*HID*HID,hout_b+(size_t)idx*HID,obuf,KT_HID,HID);
    hgt_combine_kernel<<<divup((int)NH,256),256,0,stream>>>(obuf,hcur,hskip,idx,res,(int)NH);
  };

  // ---- pipeline ----
  packA(x,xpack,KT_IN,IN_DIM);
  // layer 0: h = lrelu(bn0(gat0(x)))
  gat_layer(xpack,KT_IN,gat0_W,gat0_att,gat0_b,gout);
  bn(gout,0,nullptr,nullptr,hcur,1);
  for(int i=0;i<3;++i){
    int li=i+1;
    packA(hcur,apackH,KT_HID,HID);
    gat_layer(apackH,KT_HID,
              gat_W  +(size_t)i*3*HID*HID,
              gat_att+(size_t)i*3*2*GHEADS*GD,
              gat_b  +(size_t)i*3*HID, gout);
    const float* addB=nullptr;
    if(li>=2){ // input injection: inj = x @ proj_W[li-2] + proj_b
      gemm(xpack,proj_W+(size_t)(li-2)*IN_DIM*HID,proj_b+(size_t)(li-2)*HID,injbuf,KT_IN,HID);
      addB=injbuf;
    }
    bn(gout,1+i,hcur,addB,hcur,1);   // h = lrelu(h + bn(gout) [+ inj])
    if(li==1){
      hgt(0,vmbuf);
      bn(vmbuf,4,hcur,nullptr,hcur,1);  // h = lrelu(h + bn4(hgt0))
    }
  }
  hgt(1,vmbuf);
  bn(vmbuf,5,hcur,nullptr,hcur,1);      // h = lrelu(h + bn5(hgt1))
  lin_kernel<<<divup(N_NODES,256),256,0,stream>>>(hcur,lin_W,lin_b,(float*)d_out);
}
